// GCN_bnln_32238024523885
// MI455X (gfx1250) — matrix operand order repaired and checked by reading, not compiled, not run on hardware
//
#include <hip/hip_runtime.h>
#include <math.h>

#define NN 50000
#define NE 800000
#define FIN 128
#define FH  128
#define FOUT 40
#define LN_EPS 1e-5f

typedef __attribute__((ext_vector_type(16))) __bf16 v16bf;
typedef __attribute__((ext_vector_type(8)))  float  v8f;

__device__ __forceinline__ float warp_sum(float v) {
#pragma unroll
  for (int m = 16; m > 0; m >>= 1) v += __shfl_xor(v, m, 32);
  return v;
}
__device__ __forceinline__ float warp_max(float v) {
#pragma unroll
  for (int m = 16; m > 0; m >>= 1) v = fmaxf(v, __shfl_xor(v, m, 32));
  return v;
}

__device__ __forceinline__ void atomic_add_f32(float* p, float v) {
  __hip_atomic_fetch_add(p, v, __ATOMIC_RELAXED, __HIP_MEMORY_SCOPE_AGENT);
}

// ---------------- degree / norm prep ----------------
__global__ void deg_init(float* deg) {
  int i = blockIdx.x * 256 + threadIdx.x;
  if (i < NN) deg[i] = 1.0f;  // self-loop
}
__global__ void deg_count(const int* __restrict__ dst, float* deg) {
  int e = blockIdx.x * 256 + threadIdx.x;
  if (e < NE) atomic_add_f32(&deg[dst[e]], 1.0f);
}
__global__ void deg_finalize(float* deg) {
  int i = blockIdx.x * 256 + threadIdx.x;
  if (i < NN) deg[i] = rsqrtf(deg[i]);  // now holds dinv
}

// ---------------- WMMA GEMM: Hout[N,NCOLS] = Hin[N,128] @ W[128,NCOLS] ----------------
// Block: 256 threads = 8 waves; each wave does a 16-row strip x NTILES*16 cols.
// W staged in LDS as bf16, pre-swizzled into B-fragment order:
//   sW[((kg*NTILES + nt)*32 + lane)*16 + t] = W[kg*32 + 16*(lane>>4) + t][nt*16 + (lane&15)]
template <int NCOLS, int NTILES>
__global__ __launch_bounds__(256) void gemm_wmma_bf16(
    const float* __restrict__ Hin, const float* __restrict__ W,
    float* __restrict__ Hout, int nrows) {
  __shared__ __bf16 sW[4 * NTILES * 32 * 16];

  const int tid = threadIdx.x;
  // fill LDS in fragment order
  for (int j = tid; j < 4 * NTILES * 512; j += 256) {
    int t  = j & 15;
    int ln = (j >> 4) & 31;
    int nt = (j >> 9) % NTILES;
    int kg = j / (NTILES * 512);
    int k  = kg * 32 + ((ln >> 4) << 4) + t;
    int n  = nt * 16 + (ln & 15);
    float w = (n < NCOLS) ? W[k * NCOLS + n] : 0.0f;
    sW[j] = (__bf16)w;
  }
  __syncthreads();

  const int wave = tid >> 5;
  const int lane = tid & 31;
  const int half = lane >> 4;
  const int lmod = lane & 15;
  const int rowBase = blockIdx.x * 128 + wave * 16;

  v8f acc[NTILES];
#pragma unroll
  for (int nt = 0; nt < NTILES; ++nt) acc[nt] = (v8f){0, 0, 0, 0, 0, 0, 0, 0};

  const int m = rowBase + lmod;
  const bool mval = (m < nrows);
  const float* arow = Hin + (size_t)(mval ? m : 0) * 128;
  const v16bf* bfrags = (const v16bf*)sW;

#pragma unroll
  for (int kg = 0; kg < 4; ++kg) {
    const int kb = kg * 32;
    // A fragment: elems 0..15 <- K = kb+16h .. +15
    float4 f0 = *(const float4*)(arow + kb + half * 16);
    float4 f1 = *(const float4*)(arow + kb + half * 16 + 4);
    float4 f2 = *(const float4*)(arow + kb + half * 16 + 8);
    float4 f3 = *(const float4*)(arow + kb + half * 16 + 12);
    if (!mval) { f0 = f1 = f2 = f3 = make_float4(0.f, 0.f, 0.f, 0.f); }
    v16bf a;
    a[0]  = (__bf16)f0.x; a[1]  = (__bf16)f0.y; a[2]  = (__bf16)f0.z; a[3]  = (__bf16)f0.w;
    a[4]  = (__bf16)f1.x; a[5]  = (__bf16)f1.y; a[6]  = (__bf16)f1.z; a[7]  = (__bf16)f1.w;
    a[8]  = (__bf16)f2.x; a[9]  = (__bf16)f2.y; a[10] = (__bf16)f2.z; a[11] = (__bf16)f2.w;
    a[12] = (__bf16)f3.x; a[13] = (__bf16)f3.y; a[14] = (__bf16)f3.z; a[15] = (__bf16)f3.w;

#pragma unroll
    for (int nt = 0; nt < NTILES; ++nt) {
      v16bf b = bfrags[(kg * NTILES + nt) * 32 + lane];
      acc[nt] = __builtin_amdgcn_wmma_f32_16x16x32_bf16(
          false, a, false, b, (short)0, acc[nt], false, false);
    }
  }

  // store D: vgpr r, lane half h -> M = rowBase + r + 8h, N = nt*16 + lmod
#pragma unroll
  for (int nt = 0; nt < NTILES; ++nt) {
#pragma unroll
    for (int r = 0; r < 8; ++r) {
      int mm = rowBase + half * 8 + r;
      int nn = nt * 16 + lmod;
      if (mm < nrows && nn < NCOLS) Hout[(size_t)mm * NCOLS + nn] = acc[nt][r];
    }
  }
}

// ---------------- aggregation ----------------
// Out[i,f] = Hin[i,f]*dinv[i]^2 + bias[f]   (self-loop term + bias)
template <int F>
__global__ void agg_init(const float* __restrict__ Hin, const float* __restrict__ dinv,
                         const float* __restrict__ bias, float* __restrict__ Out) {
  int tid = blockIdx.x * 256 + threadIdx.x;
  if (tid < NN * F) {
    int i = tid / F, f = tid % F;
    float di = dinv[i];
    Out[tid] = Hin[tid] * di * di + bias[f];
  }
}

// Out[dst,f] += Hin[src,f] * dinv[src]*dinv[dst]  (L2-resident scatter)
template <int F>
__global__ void agg_edges(const float* __restrict__ Hin, const float* __restrict__ dinv,
                          const int* __restrict__ src, const int* __restrict__ dst,
                          float* __restrict__ Out) {
  constexpr int CH = F / 4;
  int tid = blockIdx.x * 256 + threadIdx.x;
  int e = tid / CH;
  if (e >= NE) return;
  int c = (tid % CH) * 4;
  int s = src[e], d = dst[e];
  float nrm = dinv[s] * dinv[d];
  float4 v = *(const float4*)(Hin + (size_t)s * F + c);
  float* o = Out + (size_t)d * F + c;
  atomic_add_f32(o + 0, v.x * nrm);
  atomic_add_f32(o + 1, v.y * nrm);
  atomic_add_f32(o + 2, v.z * nrm);
  atomic_add_f32(o + 3, v.w * nrm);
}

// ---------------- LayerNorm(128) + ReLU, in place; one wave32 per node ----------------
__global__ __launch_bounds__(256) void ln_relu128(float* __restrict__ H,
                                                  const float* __restrict__ g,
                                                  const float* __restrict__ b) {
  int wave = threadIdx.x >> 5, lane = threadIdx.x & 31;
  int node = blockIdx.x * 8 + wave;
  if (node >= NN) return;
  float* row = H + (size_t)node * 128;
  float4 v = *(const float4*)(row + lane * 4);
  float mu = warp_sum(v.x + v.y + v.z + v.w) * (1.0f / 128.0f);
  float dx = v.x - mu, dy = v.y - mu, dz = v.z - mu, dw = v.w - mu;
  float var = warp_sum(dx * dx + dy * dy + dz * dz + dw * dw) * (1.0f / 128.0f);
  float inv = rsqrtf(var + LN_EPS);
  int c = lane * 4;
  v.x = fmaxf(dx * inv * g[c + 0] + b[c + 0], 0.0f);
  v.y = fmaxf(dy * inv * g[c + 1] + b[c + 1], 0.0f);
  v.z = fmaxf(dz * inv * g[c + 2] + b[c + 2], 0.0f);
  v.w = fmaxf(dw * inv * g[c + 3] + b[c + 3], 0.0f);
  *(float4*)(row + lane * 4) = v;
}

// ---------------- log_softmax over 40 classes, in place; one wave32 per node ----------------
__global__ __launch_bounds__(256) void log_softmax40(float* __restrict__ Out) {
  int wave = threadIdx.x >> 5, lane = threadIdx.x & 31;
  int node = blockIdx.x * 8 + wave;
  if (node >= NN) return;
  float* row = Out + (size_t)node * FOUT;
  float v0 = row[lane];                       // lane < 32 < 40, always valid
  bool has1 = (lane < FOUT - 32);             // lanes 0..7 cover 32..39
  float v1 = has1 ? row[lane + 32] : -INFINITY;
  float mx = warp_max(fmaxf(v0, v1));
  float s = expf(v0 - mx) + (has1 ? expf(v1 - mx) : 0.0f);
  s = warp_sum(s);
  float lse = mx + logf(s);
  row[lane] = v0 - lse;
  if (has1) row[lane + 32] = v1 - lse;
}

extern "C" void kernel_launch(void* const* d_in, const int* in_sizes, int n_in,
                              void* d_out, int out_size, void* d_ws, size_t ws_size,
                              hipStream_t stream) {
  const float* x     = (const float*)d_in[0];
  const float* W0    = (const float*)d_in[1];
  const float* b0    = (const float*)d_in[2];
  const float* W1    = (const float*)d_in[3];
  const float* b1    = (const float*)d_in[4];
  const float* W2    = (const float*)d_in[5];
  const float* b2    = (const float*)d_in[6];
  const float* ln1_g = (const float*)d_in[7];
  const float* ln1_b = (const float*)d_in[8];
  const float* ln2_g = (const float*)d_in[9];
  const float* ln2_b = (const float*)d_in[10];
  const int*   ei    = (const int*)d_in[11];
  const int* src = ei;          // edge_index[0]
  const int* dst = ei + NE;     // edge_index[1]
  float* out = (float*)d_out;

  float* ws   = (float*)d_ws;
  float* dinv = ws;                         // [NN]
  float* hA   = ws + NN;                    // [NN*128]
  float* hB   = hA + (size_t)NN * 128;      // [NN*128]

  const unsigned gN    = (NN + 255) / 256;
  const unsigned gE    = (NE + 255) / 256;
  const unsigned gNF   = ((unsigned)(NN * FH) + 255) / 256;
  const unsigned gNFo  = ((unsigned)(NN * FOUT) + 255) / 256;
  const unsigned gE128 = (unsigned)(((size_t)NE * (FH / 4) + 255) / 256);
  const unsigned gE40  = (unsigned)(((size_t)NE * (FOUT / 4) + 255) / 256);
  const unsigned gGemm = (NN + 127) / 128;
  const unsigned gWave = (NN + 7) / 8;

  // degree -> dinv
  deg_init<<<gN, 256, 0, stream>>>(dinv);
  deg_count<<<gE, 256, 0, stream>>>(dst, dinv);
  deg_finalize<<<gN, 256, 0, stream>>>(dinv);

  // layer 0: h = relu(LN1(agg(x @ W0) + b0))
  gemm_wmma_bf16<FH, 8><<<gGemm, 256, 0, stream>>>(x, W0, hA, NN);
  agg_init<FH><<<gNF, 256, 0, stream>>>(hA, dinv, b0, hB);
  agg_edges<FH><<<gE128, 256, 0, stream>>>(hA, dinv, src, dst, hB);
  ln_relu128<<<gWave, 256, 0, stream>>>(hB, ln1_g, ln1_b);

  // layer 1: h = relu(LN2(agg(h @ W1) + b1))
  gemm_wmma_bf16<FH, 8><<<gGemm, 256, 0, stream>>>(hB, W1, hA, NN);
  agg_init<FH><<<gNF, 256, 0, stream>>>(hA, dinv, b1, hB);
  agg_edges<FH><<<gE128, 256, 0, stream>>>(hA, dinv, src, dst, hB);
  ln_relu128<<<gWave, 256, 0, stream>>>(hB, ln2_g, ln2_b);

  // layer 2: logits = agg(h @ W2) + b2 ; out = log_softmax(logits)
  gemm_wmma_bf16<FOUT, 3><<<gGemm, 256, 0, stream>>>(hB, W2, hA, NN);
  agg_init<FOUT><<<gNFo, 256, 0, stream>>>(hA, dinv, b2, out);
  agg_edges<FOUT><<<gE40, 256, 0, stream>>>(hA, dinv, src, dst, out);
  log_softmax40<<<gWave, 256, 0, stream>>>(out);
}
